// SlotMotionPredictor_52974126629405
// MI455X (gfx1250) — matrix operand order repaired and checked by reading, not compiled, not run on hardware
//
#include <hip/hip_runtime.h>
#include <hip/hip_fp16.h>

// MI455X / gfx1250, wave32. All dense layers -> v_wmma_f32_16x16x32_f16.
// Workspace requirement: ~64 MB.

typedef __attribute__((ext_vector_type(16))) _Float16 v16h;
typedef __attribute__((ext_vector_type(8)))  _Float16 v8h;
typedef __attribute__((ext_vector_type(4)))  _Float16 v4h;
typedef __attribute__((ext_vector_type(8)))  float    v8f;

#define DEVI __device__ __forceinline__

DEVI float gelu_f(float x) { return 0.5f * x * (1.0f + erff(x * 0.7071067811865475f)); }
DEVI float sigm_f(float x) { return 1.0f / (1.0f + expf(-x)); }

// ---------------------------------------------------------------------------
// Generic WMMA GEMM:  D[m][n] = epi( sum_k A[m][k] * Bt[n][k] + bias[n] )
// A: MxK f16 row-major.  Bt: NxK f16 row-major (i.e. weight [out][in]).
// One wave computes a 16 x (16*NT) tile.  M,K mult of 16/32; N mult of 16*NT.
// EPI: 0=none, 1=bias, 2=bias+GELU.  OUTH: 1 -> f16 out, 0 -> f32 out.
// ---------------------------------------------------------------------------
template <int NT, int EPI, int OUTH>
__global__ void gemm_wmma_k(const _Float16* __restrict__ A,
                            const _Float16* __restrict__ Bt,
                            const float* __restrict__ bias,
                            void* __restrict__ Dp,
                            int M, int N, int K)
{
    const int lane   = threadIdx.x & 31;
    const int wid    = (blockIdx.x * blockDim.x + threadIdx.x) >> 5;
    const int ntiles = N / (16 * NT);
    const int totalW = (M >> 4) * ntiles;
    if (wid >= totalW) return;                 // wave-uniform guard
    const int mt  = wid / ntiles;
    const int nt0 = (wid - mt * ntiles) * NT;
    const int r   = lane & 15;
    const size_t arow = (size_t)(mt * 16 + r) * K + ((lane >> 4) << 4);
    const int col0 = nt0 * 16 + r;

    v8f acc[NT] = {};
    for (int k0 = 0; k0 < K; k0 += 32) {
        v16h a  = *(const v16h*)(A + arow + k0);
#pragma unroll
        for (int j = 0; j < NT; ++j) {
            const _Float16* bp = Bt + (size_t)(col0 + j * 16) * K + k0 + ((lane >> 4) << 4);
            v16h b = *(const v16h*)bp;
            acc[j] = __builtin_amdgcn_wmma_f32_16x16x32_f16(false, a, false, b,
                                                            (short)0, acc[j], false, false);
        }
    }
    const int mhalf = (lane >> 4) << 3;
#pragma unroll
    for (int j = 0; j < NT; ++j) {
        const int col = col0 + j * 16;
        const float bb = (EPI >= 1) ? bias[col] : 0.0f;
#pragma unroll
        for (int g = 0; g < 8; ++g) {
            const int m  = mt * 16 + mhalf + g;
            float v = acc[j][g] + bb;
            if (EPI == 2) v = gelu_f(v);
            const size_t idx = (size_t)m * N + col;
            if (OUTH) ((_Float16*)Dp)[idx] = (_Float16)v;
            else      ((float*)Dp)[idx]    = v;
        }
    }
}

static inline int gemm_grid(int M, int N, int NT)
{
    int waves = (M / 16) * (N / (16 * NT));
    return (waves + 7) / 8;                    // 8 waves (256 threads) per block
}

// ---------------------------------------------------------------------------
// enc1: direct 3x3 conv (2->128) + GELU, output written straight into the
// im2col A-matrix layout for the enc2 GEMM:
//   A[b*16 + (y/4)*4 + (x/4)][co*16 + (y%4)*4 + (x%4)]
// ---------------------------------------------------------------------------
__global__ void enc1_k(const float* __restrict__ frames,
                       const float* __restrict__ w1, const float* __restrict__ b1,
                       _Float16* __restrict__ Aenc2, int s, int T)
{
    __shared__ float sp[2][18][18];
    __shared__ float sw[2304];
    __shared__ float sb[128];
    const int b = blockIdx.x, t = threadIdx.x;
    for (int i = t; i < 2 * 18 * 18; i += 256) ((float*)sp)[i] = 0.0f;
    for (int i = t; i < 2304; i += 256) sw[i] = w1[i];
    if (t < 128) sb[t] = b1[t];
    __syncthreads();
    {
        const float* f0 = frames + ((size_t)b * T + (s + 1)) * 256;
        const float* f1 = frames + ((size_t)b * T + s) * 256;
        const int y = t >> 4, x = t & 15;
        sp[0][y + 1][x + 1] = f0[t];
        sp[1][y + 1][x + 1] = f1[t];
    }
    __syncthreads();
    const int y = t >> 4, x = t & 15;
    float v[18];
#pragma unroll
    for (int ci = 0; ci < 2; ++ci)
#pragma unroll
        for (int ky = 0; ky < 3; ++ky)
#pragma unroll
            for (int kx = 0; kx < 3; ++kx)
                v[ci * 9 + ky * 3 + kx] = sp[ci][y + ky][x + kx];
    const size_t rowbase = (size_t)(b * 16 + (y >> 2) * 4 + (x >> 2)) * 2048
                         + (size_t)((y & 3) * 4 + (x & 3));
    for (int co = 0; co < 128; ++co) {
        float acc = sb[co];
        const float* wc = &sw[co * 18];
#pragma unroll
        for (int q = 0; q < 18; ++q) acc += v[q] * wc[q];
        Aenc2[rowbase + (size_t)co * 16] = (_Float16)gelu_f(acc);
    }
}

// ---------------------------------------------------------------------------
// Per-batch slot attention: scores = qq@kk^T via one WMMA tile, softmax over
// the 8 slots (axis=1), then upd = attn@vv via shfl broadcast (no LDS).
// One wave per batch.
// ---------------------------------------------------------------------------
__global__ void attn_upd_k(const _Float16* __restrict__ qq,
                           const _Float16* __restrict__ kkm,
                           const _Float16* __restrict__ vv,
                           _Float16* __restrict__ upd, int Bq)
{
    const int lane = threadIdx.x & 31;
    const int b = (blockIdx.x * blockDim.x + threadIdx.x) >> 5;
    if (b >= Bq) return;
    const int r = lane & 15;
    const size_t aoff = (size_t)(b * 8 + (r & 7)) * 128 + ((lane >> 4) << 4);
    const size_t boff = (size_t)(b * 16 + r) * 128 + ((lane >> 4) << 4);
    v8f sco = {};
#pragma unroll
    for (int k0 = 0; k0 < 128; k0 += 32) {
        v16h a  = *(const v16h*)(qq + aoff + k0);
        v16h bf = *(const v16h*)(kkm + boff + k0);
        sco = __builtin_amdgcn_wmma_f32_16x16x32_f16(false, a, false, bf, (short)0, sco, false, false);
    }
    // lanes 0..15 hold column n=lane, rows m=0..7 in regs 0..7 -> softmax over slots
    const float scale = 0.08838834764831845f;  // 1/sqrt(128)
    float e[8];
    float mx = -1e30f;
#pragma unroll
    for (int g = 0; g < 8; ++g) { e[g] = sco[g] * scale; mx = fmaxf(mx, e[g]); }
    float sum = 0.0f;
#pragma unroll
    for (int g = 0; g < 8; ++g) { e[g] = expf(e[g] - mx); sum += e[g]; }
    const float inv = 1.0f / sum;
#pragma unroll
    for (int g = 0; g < 8; ++g) e[g] *= inv;   // attn[k=g][n=lane], valid lanes 0..15
    // upd[k][d] = sum_n attn[k][n] * vv[b][n][d]; lane owns d = lane*4 .. lane*4+3
    float acc[8][4];
#pragma unroll
    for (int g = 0; g < 8; ++g)
#pragma unroll
        for (int j = 0; j < 4; ++j) acc[g][j] = 0.0f;
    for (int n = 0; n < 16; ++n) {
        v4h vl = *(const v4h*)(vv + (size_t)(b * 16 + n) * 128 + lane * 4);
        float vf0 = (float)vl[0], vf1 = (float)vl[1], vf2 = (float)vl[2], vf3 = (float)vl[3];
#pragma unroll
        for (int g = 0; g < 8; ++g) {
            const float a_ = __shfl(e[g], n);
            acc[g][0] += a_ * vf0; acc[g][1] += a_ * vf1;
            acc[g][2] += a_ * vf2; acc[g][3] += a_ * vf3;
        }
    }
#pragma unroll
    for (int g = 0; g < 8; ++g) {
        v4h o;
        o[0] = (_Float16)acc[g][0]; o[1] = (_Float16)acc[g][1];
        o[2] = (_Float16)acc[g][2]; o[3] = (_Float16)acc[g][3];
        *(v4h*)(upd + (size_t)(b * 8 + g) * 128 + lane * 4) = o;
    }
}

// GRU gating (biases already folded into gi/gh by GEMM epilogue)
__global__ void gru_gate_k(const float* __restrict__ gi, const float* __restrict__ gh,
                           float* __restrict__ slots, _Float16* __restrict__ slots_h, int n)
{
    const int i = blockIdx.x * blockDim.x + threadIdx.x;
    if (i >= n) return;
    const int row = i >> 7, d = i & 127;
    const float* gr = gi + (size_t)row * 384;
    const float* hr = gh + (size_t)row * 384;
    const float r  = sigm_f(gr[d] + hr[d]);
    const float z  = sigm_f(gr[128 + d] + hr[128 + d]);
    const float nn = tanhf(gr[256 + d] + r * hr[256 + d]);
    const float h  = slots[i];
    const float o  = (1.0f - z) * nn + z * h;
    slots[i] = o;
    slots_h[i] = (_Float16)o;
}

// motion head stage 2: (4096 x 32) -> tanh(.) * 4 -> (4096 x 2)
__global__ void mot2_k(const float* __restrict__ m1, const float* __restrict__ w2,
                       const float* __restrict__ b2, float* __restrict__ motion, int rows)
{
    const int row = blockIdx.x * blockDim.x + threadIdx.x;
    if (row >= rows) return;
    float a0 = b2[0], a1 = b2[1];
    const float* m = m1 + (size_t)row * 32;
#pragma unroll
    for (int j = 0; j < 32; ++j) { const float x = m[j]; a0 += x * w2[j]; a1 += x * w2[32 + j]; }
    motion[row * 2 + 0] = tanhf(a0) * 4.0f;
    motion[row * 2 + 1] = tanhf(a1) * 4.0f;
}

// Decoder: transposed-conv upsample (4x4 -> 16x16, stride 4), softmax over
// slots, bilinear warp, mask-weighted sum.  One block per batch image.
__global__ void warp_k(const float* __restrict__ frames, const float* __restrict__ ml,
                       const float* __restrict__ upw, const float* __restrict__ upb,
                       const float* __restrict__ motion,
                       float* __restrict__ warped, float* __restrict__ diff, int s, int T)
{
    __shared__ float sup[1024];   // up_w[ki][ko][ry*4+rx]
    __shared__ float sbb[8];
    __shared__ float simg[256];
    const int b = blockIdx.x, t = threadIdx.x;
    for (int i = t; i < 1024; i += 256) sup[i] = upw[i];
    if (t < 8) sbb[t] = upb[t];
    simg[t] = frames[((size_t)b * T + (s + 1)) * 256 + t];
    __syncthreads();
    const int Y = t >> 4, X = t & 15;
    const int p = (Y >> 2) * 4 + (X >> 2);
    const int rr = (Y & 3) * 4 + (X & 3);
    float lg[8];
#pragma unroll
    for (int ko = 0; ko < 8; ++ko) lg[ko] = sbb[ko];
#pragma unroll
    for (int ki = 0; ki < 8; ++ki) {
        const float m = ml[(size_t)(b * 8 + ki) * 16 + p];
#pragma unroll
        for (int ko = 0; ko < 8; ++ko) lg[ko] += m * sup[(ki * 8 + ko) * 16 + rr];
    }
    float mx = lg[0];
#pragma unroll
    for (int k = 1; k < 8; ++k) mx = fmaxf(mx, lg[k]);
    float sum = 0.0f;
#pragma unroll
    for (int k = 0; k < 8; ++k) { lg[k] = expf(lg[k] - mx); sum += lg[k]; }
    const float inv = 1.0f / sum;
    const float gx = -1.0f + 2.0f * X / 15.0f;
    const float gy = -1.0f + 2.0f * Y / 15.0f;
    float w = 0.0f;
#pragma unroll
    for (int k = 0; k < 8; ++k) {
        const float fx = motion[(b * 8 + k) * 2 + 0] * 0.125f;  // *2/16
        const float fy = motion[(b * 8 + k) * 2 + 1] * 0.125f;
        const float sgx = fminf(fmaxf(gx + fx, -1.0f), 1.0f);
        const float sgy = fminf(fmaxf(gy + fy, -1.0f), 1.0f);
        const float ix = (sgx + 1.0f) * 0.5f * 15.0f;
        const float iy = (sgy + 1.0f) * 0.5f * 15.0f;
        const float x0f = floorf(ix), y0f = floorf(iy);
        const float wx = ix - x0f, wy = iy - y0f;
        const int x0 = min(max((int)x0f, 0), 15), x1 = min((int)x0f + 1, 15);
        const int y0 = min(max((int)y0f, 0), 15), y1 = min((int)y0f + 1, 15);
        const float v00 = simg[y0 * 16 + x0], v01 = simg[y0 * 16 + x1];
        const float v10 = simg[y1 * 16 + x0], v11 = simg[y1 * 16 + x1];
        const float wk = (1.0f - wy) * ((1.0f - wx) * v00 + wx * v01)
                       + wy * ((1.0f - wx) * v10 + wx * v11);
        w += (lg[k] * inv) * wk;
    }
    warped[b * 256 + t] = w;
    diff[b * 256 + t] = w - simg[t];
}

// Fused residual: conv 1->64 (3x3,pad1) + GELU in LDS, conv 64->1, tanh*0.1,
// pred = clip(warped + res).  One block per batch image.
__global__ void res_k(const float* __restrict__ diff, const float* __restrict__ warped,
                      const float* __restrict__ w1, const float* __restrict__ b1,
                      const float* __restrict__ w2, const float* __restrict__ b2,
                      float* __restrict__ out, int s)
{
    __shared__ float    sd[18][18];
    __shared__ _Float16 sr[64][18][18];
    __shared__ float    sw1[576];
    __shared__ float    sw2[576];
    __shared__ float    sb1[64];
    const int b = blockIdx.x, t = threadIdx.x;
    for (int i = t; i < 324; i += 256) ((float*)sd)[i] = 0.0f;
    for (int i = t; i < 64 * 324; i += 256) ((_Float16*)sr)[i] = (_Float16)0.0f;
    for (int i = t; i < 576; i += 256) { sw1[i] = w1[i]; sw2[i] = w2[i]; }
    if (t < 64) sb1[t] = b1[t];
    __syncthreads();
    const int y = t >> 4, x = t & 15;
    sd[y + 1][x + 1] = diff[b * 256 + t];
    __syncthreads();
    float v[9];
#pragma unroll
    for (int ky = 0; ky < 3; ++ky)
#pragma unroll
        for (int kx = 0; kx < 3; ++kx) v[ky * 3 + kx] = sd[y + ky][x + kx];
    for (int c = 0; c < 64; ++c) {
        float a = sb1[c];
#pragma unroll
        for (int q = 0; q < 9; ++q) a += v[q] * sw1[c * 9 + q];
        sr[c][y + 1][x + 1] = (_Float16)gelu_f(a);
    }
    __syncthreads();
    float acc = b2[0];
    for (int c = 0; c < 64; ++c) {
#pragma unroll
        for (int ky = 0; ky < 3; ++ky)
#pragma unroll
            for (int kx = 0; kx < 3; ++kx)
                acc += (float)sr[c][y + ky][x + kx] * sw2[c * 9 + ky * 3 + kx];
    }
    const float res = tanhf(acc) * 0.1f;
    float pred = warped[b * 256 + t] + res;
    pred = fminf(fmaxf(pred, 0.0f), 1.0f);
    out[((size_t)b * 16 + s) * 256 + t] = pred;
}

// ------------------------- prep kernels (once per call) --------------------
__global__ void cvt_f16_k(const float* __restrict__ src, _Float16* __restrict__ dst, int n)
{
    const int i = blockIdx.x * blockDim.x + threadIdx.x;
    if (i < n) dst[i] = (_Float16)src[i];
}

// Fold mean over c (128 groups) into mask weights: wred[p][d], bred[p]
__global__ void maskred_k(const float* __restrict__ mask_w, const float* __restrict__ mask_b,
                          _Float16* __restrict__ wred, float* __restrict__ bred)
{
    const int i = blockIdx.x * blockDim.x + threadIdx.x;
    if (i < 2048) {
        const int p = i >> 7, d = i & 127;
        float sum = 0.0f;
        for (int c = 0; c < 128; ++c) sum += mask_w[(size_t)(c * 16 + p) * 128 + d];
        wred[p * 128 + d] = (_Float16)(sum * (1.0f / 128.0f));
    }
    if (i < 16) {
        float sum = 0.0f;
        for (int c = 0; c < 128; ++c) sum += mask_b[c * 16 + i];
        bred[i] = sum * (1.0f / 128.0f);
    }
}

__global__ void slots_init_k(const float* __restrict__ mu, float* __restrict__ slots,
                             _Float16* __restrict__ slots_h, int n)
{
    const int i = blockIdx.x * blockDim.x + threadIdx.x;
    if (i >= n) return;
    const float v = mu[i & 1023];
    slots[i] = v;
    slots_h[i] = (_Float16)v;
}

// ---------------------------------------------------------------------------
extern "C" void kernel_launch(void* const* d_in, const int* in_sizes, int n_in,
                              void* d_out, int out_size, void* d_ws, size_t ws_size,
                              hipStream_t stream)
{
    (void)in_sizes; (void)n_in; (void)out_size; (void)ws_size;
    const float* frames    = (const float*)d_in[0];
    const float* enc1_w    = (const float*)d_in[1];
    const float* enc1_b    = (const float*)d_in[2];
    const float* enc2_w    = (const float*)d_in[3];
    const float* enc2_b    = (const float*)d_in[4];
    const float* slot_mu   = (const float*)d_in[5];
    const float* to_slot_w = (const float*)d_in[6];
    const float* to_slot_b = (const float*)d_in[7];
    const float* q_w       = (const float*)d_in[8];
    const float* k_w       = (const float*)d_in[9];
    const float* v_w       = (const float*)d_in[10];
    const float* gru_w_ih  = (const float*)d_in[11];
    const float* gru_w_hh  = (const float*)d_in[12];
    const float* gru_b_ih  = (const float*)d_in[13];
    const float* gru_b_hh  = (const float*)d_in[14];
    const float* mot1_w    = (const float*)d_in[15];
    const float* mot1_b    = (const float*)d_in[16];
    const float* mot2_w    = (const float*)d_in[17];
    const float* mot2_b    = (const float*)d_in[18];
    const float* mask_w    = (const float*)d_in[19];
    const float* mask_b    = (const float*)d_in[20];
    const float* up_w      = (const float*)d_in[21];
    const float* up_b      = (const float*)d_in[22];
    const float* res1_w    = (const float*)d_in[23];
    const float* res1_b    = (const float*)d_in[24];
    const float* res2_w    = (const float*)d_in[25];
    const float* res2_b    = (const float*)d_in[26];
    float* out = (float*)d_out;

    const int T = 18, STEPS = 16;
    const int M2 = 8192;   // B*16 feature rows
    const int MS = 4096;   // B*8 slot rows

    char* ws = (char*)d_ws;
    size_t off = 0;
    auto alloc = [&](size_t bytes) -> void* {
        void* p = ws + off;
        off = (off + bytes + 255) & ~(size_t)255;
        return p;
    };
    _Float16* enc2_wh  = (_Float16*)alloc((size_t)262144 * 2);
    _Float16* toslot_h = (_Float16*)alloc((size_t)16384 * 2);
    _Float16* q_h      = (_Float16*)alloc((size_t)16384 * 2);
    _Float16* k_h      = (_Float16*)alloc((size_t)16384 * 2);
    _Float16* v_h      = (_Float16*)alloc((size_t)16384 * 2);
    _Float16* wih_h    = (_Float16*)alloc((size_t)49152 * 2);
    _Float16* whh_h    = (_Float16*)alloc((size_t)49152 * 2);
    _Float16* mot1_h   = (_Float16*)alloc((size_t)4096 * 2);
    _Float16* mlw_h    = (_Float16*)alloc((size_t)2048 * 2);
    float*    mlb      = (float*)alloc((size_t)16 * 4);
    float*    slots    = (float*)alloc((size_t)MS * 128 * 4);
    _Float16* slots_h  = (_Float16*)alloc((size_t)MS * 128 * 2);
    _Float16* Aenc2    = (_Float16*)alloc((size_t)M2 * 2048 * 2);
    _Float16* feat2    = (_Float16*)alloc((size_t)M2 * 128 * 2);
    _Float16* fsb      = (_Float16*)alloc((size_t)M2 * 128 * 2);
    _Float16* kkb      = (_Float16*)alloc((size_t)M2 * 128 * 2);
    _Float16* vvb      = (_Float16*)alloc((size_t)M2 * 128 * 2);
    _Float16* qqb      = (_Float16*)alloc((size_t)MS * 128 * 2);
    _Float16* updb     = (_Float16*)alloc((size_t)MS * 128 * 2);
    float*    gib      = (float*)alloc((size_t)MS * 384 * 4);
    float*    ghb      = (float*)alloc((size_t)MS * 384 * 4);
    float*    m1b      = (float*)alloc((size_t)MS * 32 * 4);
    float*    motion   = (float*)alloc((size_t)MS * 2 * 4);
    float*    mlbuf    = (float*)alloc((size_t)MS * 16 * 4);
    float*    warped   = (float*)alloc((size_t)512 * 256 * 4);
    float*    diffb    = (float*)alloc((size_t)512 * 256 * 4);

    // ---- one-time prep: weight conversion + mask-mean folding + slot init
    cvt_f16_k<<<1024, 256, 0, stream>>>(enc2_w,    enc2_wh,  262144);
    cvt_f16_k<<<64,   256, 0, stream>>>(to_slot_w, toslot_h, 16384);
    cvt_f16_k<<<64,   256, 0, stream>>>(q_w,       q_h,      16384);
    cvt_f16_k<<<64,   256, 0, stream>>>(k_w,       k_h,      16384);
    cvt_f16_k<<<64,   256, 0, stream>>>(v_w,       v_h,      16384);
    cvt_f16_k<<<192,  256, 0, stream>>>(gru_w_ih,  wih_h,    49152);
    cvt_f16_k<<<192,  256, 0, stream>>>(gru_w_hh,  whh_h,    49152);
    cvt_f16_k<<<16,   256, 0, stream>>>(mot1_w,    mot1_h,   4096);
    maskred_k<<<8, 256, 0, stream>>>(mask_w, mask_b, mlw_h, mlb);
    slots_init_k<<<2048, 256, 0, stream>>>(slot_mu, slots, slots_h, MS * 128);

    // ---- sequential scan over 16 steps
    for (int s = 0; s < STEPS; ++s) {
        enc1_k<<<512, 256, 0, stream>>>(frames, enc1_w, enc1_b, Aenc2, s, T);
        // enc2: (8192 x 2048) x (2048 x 128), bias + GELU -> f16
        gemm_wmma_k<4, 2, 1><<<gemm_grid(M2, 128, 4), 256, 0, stream>>>(
            Aenc2, enc2_wh, enc2_b, feat2, M2, 128, 2048);
        // fs = feat2 @ to_slot_w.T + b
        gemm_wmma_k<4, 1, 1><<<gemm_grid(M2, 128, 4), 256, 0, stream>>>(
            feat2, toslot_h, to_slot_b, fsb, M2, 128, 128);
        // kk, vv
        gemm_wmma_k<4, 0, 1><<<gemm_grid(M2, 128, 4), 256, 0, stream>>>(
            fsb, k_h, nullptr, kkb, M2, 128, 128);
        gemm_wmma_k<4, 0, 1><<<gemm_grid(M2, 128, 4), 256, 0, stream>>>(
            fsb, v_h, nullptr, vvb, M2, 128, 128);
        // 3 slot-attention + GRU iterations
        for (int it = 0; it < 3; ++it) {
            gemm_wmma_k<4, 0, 1><<<gemm_grid(MS, 128, 4), 256, 0, stream>>>(
                slots_h, q_h, nullptr, qqb, MS, 128, 128);
            attn_upd_k<<<64, 256, 0, stream>>>(qqb, kkb, vvb, updb, 512);
            gemm_wmma_k<4, 1, 0><<<gemm_grid(MS, 384, 4), 256, 0, stream>>>(
                updb, wih_h, gru_b_ih, gib, MS, 384, 128);
            gemm_wmma_k<4, 1, 0><<<gemm_grid(MS, 384, 4), 256, 0, stream>>>(
                slots_h, whh_h, gru_b_hh, ghb, MS, 384, 128);
            gru_gate_k<<<2048, 256, 0, stream>>>(gib, ghb, slots, slots_h, MS * 128);
        }
        // motion head
        gemm_wmma_k<2, 2, 0><<<gemm_grid(MS, 32, 2), 256, 0, stream>>>(
            slots_h, mot1_h, mot1_b, m1b, MS, 32, 128);
        mot2_k<<<16, 256, 0, stream>>>(m1b, mot2_w, mot2_b, motion, MS);
        // mask logits with mean folded into the weights
        gemm_wmma_k<1, 1, 0><<<gemm_grid(MS, 16, 1), 256, 0, stream>>>(
            slots_h, mlw_h, mlb, mlbuf, MS, 16, 128);
        // decoder: upsample + softmax + warp, then fused residual convs
        warp_k<<<512, 256, 0, stream>>>(frames, mlbuf, up_w, up_b, motion,
                                        warped, diffb, s, T);
        res_k<<<512, 256, 0, stream>>>(diffb, warped, res1_w, res1_b,
                                       res2_w, res2_b, out, s);
    }
}
